// SWALayer_53369263620126
// MI455X (gfx1250) — matrix operand order repaired and checked by reading, not compiled, not run on hardware
//
#include <hip/hip_runtime.h>
#include <cstdint>

#define BB 2
#define SS 4096
#define DD 1024
#define DDI 2048
#define HH 16
#define HDIM 64
#define WINSZ 256

typedef __bf16 bf16;
typedef __attribute__((ext_vector_type(16))) __bf16 v16bf;
typedef __attribute__((ext_vector_type(8)))  __bf16 v8bf;
typedef __attribute__((ext_vector_type(4)))  __bf16 v4bf;
typedef __attribute__((ext_vector_type(2)))  __bf16 v2bf;
typedef __attribute__((ext_vector_type(2)))  float  vf2;
typedef __attribute__((ext_vector_type(8)))  float  v8f;

// Let the backend pick the native f32->bf16 conversion (RNE fptrunc).
__device__ __forceinline__ bf16 f2b(float f) { return (bf16)f; }
__device__ __forceinline__ v2bf pk2(float x, float y) {
  vf2 t; t[0] = x; t[1] = y;
  return __builtin_convertvector(t, v2bf);
}

__device__ __forceinline__ v16bf cat16(v8bf a, v8bf b) {
  v16bf f;
#pragma unroll
  for (int e = 0; e < 8; ++e) { f[e] = a[e]; f[e + 8] = b[e]; }
  return f;
}

__device__ __forceinline__ v8f wmma_bf16(v16bf a, v16bf b, v8f c) {
  return __builtin_amdgcn_wmma_f32_16x16x32_bf16(false, a, false, b, (short)0, c,
                                                 false, false);
}

__device__ __forceinline__ void lds_fence() {
  __builtin_amdgcn_wave_barrier();
#if __has_builtin(__builtin_amdgcn_s_wait_dscnt)
  __builtin_amdgcn_s_wait_dscnt(0);
#else
  asm volatile("s_wait_dscnt 0" ::: "memory");
#endif
  __builtin_amdgcn_wave_barrier();
}

__device__ __forceinline__ void async_wait0() {
#if __has_builtin(__builtin_amdgcn_s_wait_asynccnt)
  __builtin_amdgcn_s_wait_asynccnt(0);
#else
  asm volatile("s_wait_asynccnt 0x0" ::: "memory");
#endif
}

// Async global->LDS copy (bf16 passthrough, no VGPR data): ASYNCcnt-tracked.
__device__ __forceinline__ void async_copy_b128(unsigned lds_addr, const void* g) {
  asm volatile("global_load_async_to_lds_b128 %0, %1, off"
               :: "v"(lds_addr), "v"((unsigned long long)(uintptr_t)g)
               : "memory");
}

// ---------------------------------------------------------------- RMSNorm
__global__ __launch_bounds__(256) void rmsnorm_kernel(const float* __restrict__ x,
                                                      const float* __restrict__ w,
                                                      bf16* __restrict__ out) {
  const int m = blockIdx.x;
  __shared__ float red[8];
  __shared__ float sscale;
  const float* xr = x + (size_t)m * DD;
  const float4 v = ((const float4*)xr)[threadIdx.x];
  float s = v.x * v.x + v.y * v.y + v.z * v.z + v.w * v.w;
#pragma unroll
  for (int off = 1; off < 32; off <<= 1) s += __shfl_xor(s, off, 32);
  if ((threadIdx.x & 31) == 0) red[threadIdx.x >> 5] = s;
  __syncthreads();
  if (threadIdx.x == 0) {
    float t = 0.f;
#pragma unroll
    for (int i = 0; i < 8; ++i) t += red[i];
    sscale = rsqrtf(t * (1.0f / DD) + 1e-5f);
  }
  __syncthreads();
  const float sc = sscale;
  const float4 wv4 = ((const float4*)w)[threadIdx.x];
  v2bf q0 = pk2(v.x * sc * wv4.x, v.y * sc * wv4.y);
  v2bf q1 = pk2(v.z * sc * wv4.z, v.w * sc * wv4.w);
  v4bf o4; o4[0] = q0[0]; o4[1] = q0[1]; o4[2] = q1[0]; o4[3] = q1[1];
  *(v4bf*)(out + (size_t)m * DD + threadIdx.x * 4) = o4;
}

// ----------------- one-time weight transpose+convert: f32 [K][N] -> bf16 [N][K]
__global__ __launch_bounds__(256) void wconvert_kernel(const float* __restrict__ w,
                                                       bf16* __restrict__ wt,
                                                       int K, int N) {
  __shared__ bf16 tile[32][33];
  const int n0 = blockIdx.x * 32;
  const int k0 = blockIdx.y * 32;
  const int tx = threadIdx.x & 31;
  const int ty = threadIdx.x >> 5;
#pragma unroll
  for (int i = 0; i < 4; ++i) {
    const int k = ty + i * 8;
    tile[k][tx] = f2b(w[(size_t)(k0 + k) * N + n0 + tx]);   // coalesced read
  }
  __syncthreads();
#pragma unroll
  for (int i = 0; i < 4; ++i) {
    const int n = ty + i * 8;
    wt[(size_t)(n0 + n) * K + k0 + tx] = tile[tx][n];       // coalesced write
  }
}

// --------------------------------------------------- bf16 WMMA GEMM (64x64 tile)
// C[M,N] = A_bf16[M,K] * Bt_bf16[N,K]^T (+resid). Both tiles async, ping-pong LDS.
__global__ __launch_bounds__(128) void gemm_bf16_kernel(
    const bf16* __restrict__ A, const bf16* __restrict__ Bt,
    const float* __restrict__ resid, float* __restrict__ C,
    bf16* __restrict__ Cb, int M, int N, int Kd) {
  __shared__ bf16 sA[2 * 64 * 40];   // ping-pong [row m][k], stride 40
  __shared__ bf16 sB[2 * 64 * 40];   // ping-pong [col n][k], stride 40
  const int tid = threadIdx.x;
  const int lane = tid & 31;
  const int wv = tid >> 5;
  const int half = lane >> 4;
  const int l = lane & 15;
  const int bm = blockIdx.y * 64;
  const int bn = blockIdx.x * 64;
  const int wm = (wv >> 1) * 32;
  const int wn = (wv & 1) * 32;
  v8f acc[2][2] = {};
  (void)M;

  const unsigned sABase = (unsigned)(uintptr_t)(void*)&sA[0];
  const unsigned sBBase = (unsigned)(uintptr_t)(void*)&sB[0];

  // 64 rows x 32 k of bf16 = 256 b128 chunks; 2 per thread, pure async copy.
  auto issueTile = [&](const bf16* src, int row0, unsigned base, int k0, int buf) {
#pragma unroll
    for (int i = 0; i < 2; ++i) {
      const int g = tid + 128 * i;
      const int r = g >> 2;
      const int c8 = (g & 3) * 8;
      async_copy_b128(base + (unsigned)(buf * 5120 + (r * 40 + c8) * 2),
                      src + (size_t)(row0 + r) * Kd + k0 + c8);
    }
  };

  issueTile(A, bm, sABase, 0, 0);
  issueTile(Bt, bn, sBBase, 0, 0);
  async_wait0();
  __syncthreads();

  const int steps = Kd / 32;
  for (int s = 0; s < steps; ++s) {
    const int buf = s & 1;
    const bf16* sAb = sA + buf * 2560;
    const bf16* sBb = sB + buf * 2560;
    if (s + 1 < steps) {              // overlap next-tile async fetch with WMMAs
      issueTile(A, bm, sABase, (s + 1) * 32, buf ^ 1);
      issueTile(Bt, bn, sBBase, (s + 1) * 32, buf ^ 1);
      if (s + 2 < steps) {            // gfx1250 global_prefetch_b8, K+2 tile
        __builtin_prefetch(A + (size_t)(bm + (tid & 63)) * Kd + (s + 2) * 32, 0, 1);
        __builtin_prefetch(Bt + (size_t)(bn + (tid & 63)) * Kd + (s + 2) * 32, 0, 1);
      }
    }

    v16bf af[2], bfv[2];
#pragma unroll
    for (int mt = 0; mt < 2; ++mt) {  // A frag: K groups 8*half and 16+8*half
      const bf16* ap = &sAb[(wm + mt * 16 + l) * 40 + 8 * half];
      af[mt] = cat16(*(const v8bf*)ap, *(const v8bf*)(ap + 16));
    }
#pragma unroll
    for (int nt = 0; nt < 2; ++nt) {  // B frag: K groups 8*half and 16+8*half
      const bf16* bp = &sBb[(wn + nt * 16 + l) * 40 + 8 * half];
      bfv[nt] = cat16(*(const v8bf*)bp, *(const v8bf*)(bp + 16));
    }
#pragma unroll
    for (int mt = 0; mt < 2; ++mt)
#pragma unroll
      for (int nt = 0; nt < 2; ++nt)
        acc[mt][nt] = wmma_bf16(af[mt], bfv[nt], acc[mt][nt]);

    async_wait0();
    __syncthreads();
  }

#pragma unroll
  for (int mt = 0; mt < 2; ++mt)
#pragma unroll
    for (int nt = 0; nt < 2; ++nt)
#pragma unroll
      for (int v = 0; v < 8; ++v) {   // C layout: row = v + 8*half, col = lane&15
        const int row = bm + wm + mt * 16 + v + 8 * half;
        const int col = bn + wn + nt * 16 + l;
        const size_t idx = (size_t)row * N + col;
        float val = acc[mt][nt][v];
        if (resid) val += resid[idx];
        if (C) C[idx] = val;
        if (Cb) Cb[idx] = f2b(val);
      }
}

// ------------------------------------- causal depthwise conv (K=4) + SiLU gate
__global__ __launch_bounds__(256) void convgate_kernel(
    const float* __restrict__ u, const float* __restrict__ gate,
    const float* __restrict__ cw, const float* __restrict__ cb,
    bf16* __restrict__ hout) {
  const size_t idx = (size_t)blockIdx.x * 256 + threadIdx.x;
  const int di = (int)(idx % DDI);
  const size_t m = idx / DDI;
  const int s = (int)(m % SS);
  float acc = cb[di];
#pragma unroll
  for (int t = 0; t < 4; ++t) {
    const int so = s - 3 + t;
    if (so >= 0) acc += cw[di * 4 + t] * u[(m - (size_t)(3 - t)) * DDI + di];
  }
  const float g = gate[idx];
  const float sig = 1.f / (1.f + __expf(-g));
  hout[idx] = f2b(g * sig * acc);
}

// -------------------------- sliding-window attention (flash, WMMA QK^T and PV)
__global__ __launch_bounds__(256) void swa_kernel(const bf16* __restrict__ qkv,
                                                  bf16* __restrict__ obuf) {
  extern __shared__ bf16 smem[];
  bf16* Kl = smem;                        // [512][72]  keys row-major (dims contiguous)
  bf16* Vt = smem + 512 * 72;             // [64][520]  V transposed (keys contiguous)
  bf16* Ps = smem + 512 * 72 + 64 * 520;  // per-wave P scratch [8][16*40]

  const int c = blockIdx.x & 15;
  const int h = (blockIdx.x >> 4) & 15;
  const int b = blockIdx.x >> 8;
  const int tid = threadIdx.x;
  const int lane = tid & 31;
  const int wv = tid >> 5;
  const int half = lane >> 4;
  const int l = lane & 15;

  const bf16* base = qkv + (size_t)b * SS * 3072;

  // ---- K tile: async global->LDS bf16 copy (ASYNCcnt), per-lane LDS scatter.
  const unsigned klBase = (unsigned)(uintptr_t)(void*)Kl;
#pragma unroll
  for (int i = 0; i < 16; ++i) {          // 512 keys x 8 b128 chunks / 256 thr
    const int g = tid + 256 * i;
    const int p = g >> 3;                 // key 0..511
    const int c8 = (g & 7) * 8;           // dim group of 8
    const int sk = c * WINSZ - WINSZ + p;
    if (sk >= 0)                          // p<256 in chunk 0 is never read
      async_copy_b128(klBase + (unsigned)(p * 144 + c8 * 2),
                      base + (size_t)sk * 3072 + 1024 + h * HDIM + c8);
  }
  // ---- V tile: transposed store (VGPR path), batched loads.
#pragma unroll
  for (int ii = 0; ii < 4; ++ii) {
    v4bf vvv[8]; int p_[8], c4_[8];
#pragma unroll
    for (int i = 0; i < 8; ++i) {
      const int g = tid + 256 * (ii * 8 + i);
      const int p = g >> 4;
      const int c4 = (g & 15) * 4;
      const int sk = c * WINSZ - WINSZ + p;
      v4bf t = {};
      if (sk >= 0) t = *(const v4bf*)(base + (size_t)sk * 3072 + 2048 + h * HDIM + c4);
      vvv[i] = t; p_[i] = p; c4_[i] = c4;
    }
#pragma unroll
    for (int i = 0; i < 8; ++i) {
      Vt[(c4_[i] + 0) * 520 + p_[i]] = vvv[i][0];
      Vt[(c4_[i] + 1) * 520 + p_[i]] = vvv[i][1];
      Vt[(c4_[i] + 2) * 520 + p_[i]] = vvv[i][2];
      Vt[(c4_[i] + 3) * 520 + p_[i]] = vvv[i][3];
    }
  }
  async_wait0();
  __syncthreads();

  const int r0 = wv * 32;                 // 32 query rows per wave (2 M-tiles)
  v16bf qf[2][2];
#pragma unroll
  for (int mt = 0; mt < 2; ++mt) {
    const int sq = c * WINSZ + r0 + mt * 16 + l;
    const bf16* qrow = base + (size_t)sq * 3072 + h * HDIM;
#pragma unroll
    for (int ks = 0; ks < 2; ++ks) {
      v8bf g0 = *(const v8bf*)(qrow + ks * 32 + 8 * half);
      v8bf g1 = *(const v8bf*)(qrow + ks * 32 + 16 + 8 * half);
      qf[mt][ks] = cat16(g0, g1);
    }
  }

  float mrow[2][8], lrow[2][8];
  v8f oacc[2][4] = {};
#pragma unroll
  for (int mt = 0; mt < 2; ++mt)
#pragma unroll
    for (int v = 0; v < 8; ++v) { mrow[mt][v] = -3.0e38f; lrow[mt][v] = 0.f; }

  const int jstart = (c == 0) ? 8 : 0;
  const int jend = min(16, ((r0 + 31 + WINSZ) >> 5) + 1);
  const float sscale = 0.125f;            // 1/sqrt(64)

  for (int j = jstart; j < jend; ++j) {
    const int p0 = j * 32;
    v8f sacc[2][2] = {};
#pragma unroll
    for (int ks = 0; ks < 2; ++ks) {
      v16bf kf[2];
#pragma unroll
      for (int kt = 0; kt < 2; ++kt) {
        const bf16* kp = &Kl[(p0 + kt * 16 + l) * 72 + ks * 32 + 8 * half];
        kf[kt] = cat16(*(const v8bf*)kp, *(const v8bf*)(kp + 16));
      }
#pragma unroll
      for (int mt = 0; mt < 2; ++mt)
#pragma unroll
        for (int kt = 0; kt < 2; ++kt)
          sacc[mt][kt] = wmma_bf16(qf[mt][ks], kf[kt], sacc[mt][kt]);
    }

#pragma unroll
    for (int mt = 0; mt < 2; ++mt) {
      const int ibase = r0 + mt * 16 + 8 * half;
      float bm[8], rs[8];
#pragma unroll
      for (int v = 0; v < 8; ++v) {
        const int iq = ibase + v;
#pragma unroll
        for (int kt = 0; kt < 2; ++kt) {
          const int p = p0 + kt * 16 + l;
          const int diff = iq + WINSZ - p;           // query_global - key_global
          const bool valid = (diff >= 0) && (diff < WINSZ);
          sacc[mt][kt][v] = valid ? sacc[mt][kt][v] * sscale : -1.0e30f;
        }
        bm[v] = fmaxf(sacc[mt][0][v], sacc[mt][1][v]);
      }
#pragma unroll
      for (int off = 1; off < 16; off <<= 1)
#pragma unroll
        for (int v = 0; v < 8; ++v)
          bm[v] = fmaxf(bm[v], __shfl_xor(bm[v], off, 32));
      float scl[8];
#pragma unroll
      for (int v = 0; v < 8; ++v) {
        const float mn = fmaxf(mrow[mt][v], bm[v]);
        scl[v] = __expf(mrow[mt][v] - mn);
        mrow[mt][v] = mn;
      }
#pragma unroll
      for (int v = 0; v < 8; ++v) {
        const float e0 = __expf(sacc[mt][0][v] - mrow[mt][v]);
        const float e1 = __expf(sacc[mt][1][v] - mrow[mt][v]);
        sacc[mt][0][v] = e0; sacc[mt][1][v] = e1;
        rs[v] = e0 + e1;
      }
#pragma unroll
      for (int off = 1; off < 16; off <<= 1)
#pragma unroll
        for (int v = 0; v < 8; ++v)
          rs[v] += __shfl_xor(rs[v], off, 32);
#pragma unroll
      for (int v = 0; v < 8; ++v) lrow[mt][v] = lrow[mt][v] * scl[v] + rs[v];
#pragma unroll
      for (int nt = 0; nt < 4; ++nt)
#pragma unroll
        for (int v = 0; v < 8; ++v) oacc[mt][nt][v] *= scl[v];

      // relayout P (C-tile layout) -> A-fragment layout via per-wave LDS scratch
      bf16* pw = Ps + wv * (16 * 40);
#pragma unroll
      for (int v = 0; v < 8; ++v) {
        const int rowloc = v + 8 * half;
        const v2bf pp = pk2(sacc[mt][0][v], sacc[mt][1][v]);
        pw[rowloc * 40 + l]      = pp[0];
        pw[rowloc * 40 + 16 + l] = pp[1];
      }
      lds_fence();
      const bf16* pr = pw + l * 40;
      v16bf pf = cat16(*(const v8bf*)(pr + 8 * half),
                       *(const v8bf*)(pr + 16 + 8 * half));
      lds_fence();
#pragma unroll
      for (int nt = 0; nt < 4; ++nt) {
        const bf16* vp = &Vt[(nt * 16 + l) * 520 + p0 + 8 * half];
        v16bf vf = cat16(*(const v8bf*)vp, *(const v8bf*)(vp + 16));
        oacc[mt][nt] = wmma_bf16(pf, vf, oacc[mt][nt]);
      }
    }
  }

#pragma unroll
  for (int mt = 0; mt < 2; ++mt)
#pragma unroll
    for (int nt = 0; nt < 4; ++nt)
#pragma unroll
      for (int v = 0; v < 8; v += 2) {
        const int s0 = c * WINSZ + r0 + mt * 16 + v + 8 * half;
        const int col = h * HDIM + nt * 16 + l;
        const v2bf pp = pk2(oacc[mt][nt][v] / lrow[mt][v],
                            oacc[mt][nt][v + 1] / lrow[mt][v + 1]);
        obuf[((size_t)(b * SS + s0)) * DD + col] = pp[0];
        obuf[((size_t)(b * SS + s0 + 1)) * DD + col] = pp[1];
      }
}

// ------------------------------------------------------------------ launcher
extern "C" void kernel_launch(void* const* d_in, const int* in_sizes, int n_in,
                              void* d_out, int out_size, void* d_ws, size_t ws_size,
                              hipStream_t stream) {
  (void)in_sizes; (void)n_in; (void)out_size; (void)ws_size;
  const float* x      = (const float*)d_in[0];
  const float* norm_w = (const float*)d_in[1];
  const float* w_up   = (const float*)d_in[2];
  const float* w_gate = (const float*)d_in[3];
  const float* w_down = (const float*)d_in[4];
  const float* conv_w = (const float*)d_in[5];
  const float* conv_b = (const float*)d_in[6];
  const float* w_qkv  = (const float*)d_in[7];
  const float* w_o    = (const float*)d_in[8];
  float* out = (float*)d_out;

  const int Mrows = BB * SS;  // 8192
  char* ws = (char*)d_ws;
  size_t off = 0;
  auto alloc = [&](size_t bytes) {
    void* p = ws + off;
    off += (bytes + 255) & ~(size_t)255;
    return p;
  };
  bf16* normed = (bf16*)alloc((size_t)Mrows * DD * 2);
  float* u     = (float*)alloc((size_t)Mrows * DDI * 4);
  float* gate  = (float*)alloc((size_t)Mrows * DDI * 4);
  bf16* hbuf   = (bf16*)alloc((size_t)Mrows * DDI * 2);
  float* x1    = (float*)alloc((size_t)Mrows * DD * 4);
  bf16* x1b    = (bf16*)alloc((size_t)Mrows * DD * 2);
  bf16* qkvb   = (bf16*)alloc((size_t)Mrows * 3 * DD * 2);
  bf16* wupT   = (bf16*)alloc((size_t)DD * DDI * 2);     // [N][K] bf16 weights
  bf16* wgateT = (bf16*)alloc((size_t)DD * DDI * 2);
  bf16* wdownT = (bf16*)alloc((size_t)DDI * DD * 2);
  bf16* wqkvT  = (bf16*)alloc((size_t)DD * 3 * DD * 2);
  bf16* woT    = (bf16*)alloc((size_t)DD * DD * 2);
  bf16* ob     = normed;  // normed is dead after up/gate GEMMs; reuse for attn out

  // one-time weight transpose+convert (bf16 [N][K])
  wconvert_kernel<<<dim3(DDI / 32, DD / 32), 256, 0, stream>>>(w_up, wupT, DD, DDI);
  wconvert_kernel<<<dim3(DDI / 32, DD / 32), 256, 0, stream>>>(w_gate, wgateT, DD, DDI);
  wconvert_kernel<<<dim3(DD / 32, DDI / 32), 256, 0, stream>>>(w_down, wdownT, DDI, DD);
  wconvert_kernel<<<dim3(3 * DD / 32, DD / 32), 256, 0, stream>>>(w_qkv, wqkvT, DD, 3 * DD);
  wconvert_kernel<<<dim3(DD / 32, DD / 32), 256, 0, stream>>>(w_o, woT, DD, DD);

  rmsnorm_kernel<<<Mrows, 256, 0, stream>>>(x, norm_w, normed);

  gemm_bf16_kernel<<<dim3(DDI / 64, Mrows / 64), 128, 0, stream>>>(
      normed, wupT, nullptr, u, nullptr, Mrows, DDI, DD);
  gemm_bf16_kernel<<<dim3(DDI / 64, Mrows / 64), 128, 0, stream>>>(
      normed, wgateT, nullptr, gate, nullptr, Mrows, DDI, DD);

  convgate_kernel<<<(Mrows * DDI) / 256, 256, 0, stream>>>(u, gate, conv_w,
                                                           conv_b, hbuf);

  gemm_bf16_kernel<<<dim3(DD / 64, Mrows / 64), 128, 0, stream>>>(
      hbuf, wdownT, x, x1, x1b, Mrows, DD, DDI);

  gemm_bf16_kernel<<<dim3(3 * DD / 64, Mrows / 64), 128, 0, stream>>>(
      x1b, wqkvT, nullptr, nullptr, qkvb, Mrows, 3 * DD, DD);

  const int shbytes = (512 * 72 + 64 * 520 + 8 * 16 * 40) * 2;  // 150528 B
  hipFuncSetAttribute(reinterpret_cast<const void*>(swa_kernel),
                      hipFuncAttributeMaxDynamicSharedMemorySize, shbytes);
  swa_kernel<<<BB * HH * (SS / WINSZ), 256, shbytes, stream>>>(qkvb, ob);

  gemm_bf16_kernel<<<dim3(DD / 64, Mrows / 64), 128, 0, stream>>>(
      ob, woT, x1, out, nullptr, Mrows, DD, DD);
}
